// MultiHeadAttention_30657476559038
// MI455X (gfx1250) — matrix operand order repaired and checked by reading, not compiled, not run on hardware
//
#include <hip/hip_runtime.h>
#include <hip/hip_bf16.h>
#include <stdint.h>

typedef __bf16 bf16;
typedef __attribute__((ext_vector_type(16))) __bf16 v16bf;
typedef __attribute__((ext_vector_type(8)))  __bf16 v8bf;
typedef __attribute__((ext_vector_type(8)))  float  v8f;

constexpr int B_ = 128, T_ = 256, D_ = 384, H_ = 6, HD_ = 64;

__device__ __forceinline__ v16bf make_v16(v8bf lo, v8bf hi) {
  return __builtin_shufflevector(lo, hi, 0,1,2,3,4,5,6,7,8,9,10,11,12,13,14,15);
}

__device__ __forceinline__ v8f wmma_bf16(v16bf a, v16bf b, v8f c) {
  // D = A(16x32 bf16) * B(32x16 bf16) + C(16x16 f32)
  return __builtin_amdgcn_wmma_f32_16x16x32_bf16(false, a, false, b, (short)0, c, false, false);
}

// Async copy of 16 bytes global -> LDS (CDNA5 GLOBAL_LOAD_ASYNC_TO_LDS_B128).
// First operand: VGPR holding LDS byte offset (low 32 bits of flat shared addr
// per the ISA LDS aperture mapping). Second: 64-bit global address. Tracked by ASYNCcnt.
__device__ __forceinline__ void async_copy16(const bf16* gsrc, bf16* ldst) {
  unsigned lds_off = (unsigned)(uintptr_t)ldst;
  asm volatile("global_load_async_to_lds_b128 %0, %1, off"
               :: "v"(lds_off), "v"(gsrc) : "memory");
}
__device__ __forceinline__ void wait_async0() {
  asm volatile("s_wait_asynccnt 0" ::: "memory");
}

// A-fragment (16x32) from row-major strip with leading dim ld, k-offset k0.
// ISA layout: lane r (r<16) holds K = 0..7 (j=0..7) and 16..23 (j=8..15);
// lane r+16 holds K = 8..15 and 24..31.  => per lane: [k0+hi*8 .. +7], [k0+16+hi*8 .. +7]
__device__ __forceinline__ v16bf a_frag(const bf16* strip, int ld, int k0) {
  const int lane = threadIdx.x & 31;
  const int r    = lane & 15;
  const int h8   = (lane >> 4) << 3;
  const bf16* row = strip + r * ld + k0 + h8;
  v8bf x0 = *(const v8bf*)(row);
  v8bf x1 = *(const v8bf*)(row + 16);
  return make_v16(x0, x1);
}

// B-fragment (32x16): B(k,c) = base[c*ld + k0 + k].
// ISA layout: lanes 0-15 hold K=0..7 and 16..23 for column c=lane; lanes 16-31 hold K=8..15 and 24..31.
__device__ __forceinline__ v16bf b_frag(const bf16* base, int ld, int k0) {
  const int lane = threadIdx.x & 31;
  const int c    = lane & 15;
  const int h8   = (lane >> 4) << 3;
  const bf16* p = base + c * ld + k0 + h8;
  v8bf x0 = *(const v8bf*)(p);
  v8bf x1 = *(const v8bf*)(p + 16);
  return make_v16(x0, x1);
}

__device__ __forceinline__ float half_max(float v) {
#pragma unroll
  for (int d = 8; d >= 1; d >>= 1) v = fmaxf(v, __shfl_xor(v, d, 32));
  return v;
}
__device__ __forceinline__ float half_sum(float v) {
#pragma unroll
  for (int d = 8; d >= 1; d >>= 1) v += __shfl_xor(v, d, 32);
  return v;
}

// ---------------- conversion / transposition kernels ----------------
__global__ void cvt_f32_bf16(const float* __restrict__ in, bf16* __restrict__ out, int n) {
  int i = blockIdx.x * 256 + threadIdx.x;
  if (i < n) out[i] = (bf16)in[i];
}

// [H,D,HD] f32 -> [H,HD,D] bf16
__global__ void cvt_tr_w(const float* __restrict__ in, bf16* __restrict__ out) {
  int i = blockIdx.x * 256 + threadIdx.x;
  if (i < H_ * D_ * HD_) {
    int h = i / (D_ * HD_);
    int r = i % (D_ * HD_);
    int d = r / HD_, k = r % HD_;
    out[((size_t)h * HD_ + k) * D_ + d] = (bf16)in[i];
  }
}

// [D,D] f32 -> transposed bf16
__global__ void cvt_tr_wp(const float* __restrict__ in, bf16* __restrict__ out) {
  int i = blockIdx.x * 256 + threadIdx.x;
  if (i < D_ * D_) {
    int r = i / D_, c = i % D_;
    out[(size_t)c * D_ + r] = (bf16)in[i];
  }
}

// ---------------- QKV projection ----------------
// grid B*T/16, block 128 (4 waves). One x-strip load per block (async -> LDS),
// then each wave computes output cols [16w,16w+16) for all H heads x {Q,K,V}.
__global__ __launch_bounds__(128) void qkv_kernel(
    const bf16* __restrict__ xb,   // [B,T,D]
    const bf16* __restrict__ Wqt,  // [H,HD,D]
    const bf16* __restrict__ Wkt,
    const bf16* __restrict__ Wvt,
    bf16* __restrict__ q,          // [B,H,T,HD], pre-scaled by 1/8
    bf16* __restrict__ k,          // [B,H,T,HD]
    bf16* __restrict__ vt)         // [B,H,HD,T] (transposed)
{
  __shared__ __align__(16) bf16 sx[16 * D_];
  const int b  = blockIdx.x / (T_ / 16);
  const int ti = blockIdx.x % (T_ / 16);
  const int t0 = ti * 16;

  const bf16* src = xb + ((size_t)b * T_ + t0) * D_;
#pragma unroll
  for (int e = threadIdx.x * 8; e < 16 * D_; e += 128 * 8)
    async_copy16(src + e, sx + e);
  wait_async0();
  __syncthreads();

  const int wave = threadIdx.x >> 5;
  const int lane = threadIdx.x & 31;
  const int cn = lane & 15, hi = lane >> 4;
  const int n0 = wave * 16;

  for (int h = 0; h < H_; ++h) {
#pragma unroll
    for (int which = 0; which < 3; ++which) {
      const bf16* Wt = (which == 0) ? Wqt : (which == 1) ? Wkt : Wvt;
      const bf16* wbase = Wt + ((size_t)h * HD_ + n0) * D_;
      v8f acc = {0.f,0.f,0.f,0.f,0.f,0.f,0.f,0.f};
#pragma unroll
      for (int kt = 0; kt < D_; kt += 32)
        acc = wmma_bf16(a_frag(sx, D_, kt), b_frag(wbase, D_, kt), acc);

      if (which == 0) {
#pragma unroll
        for (int i = 0; i < 8; ++i) {
          int m = i + hi * 8;
          q[(((size_t)b * H_ + h) * T_ + t0 + m) * HD_ + n0 + cn] = (bf16)(acc[i] * 0.125f);
        }
      } else if (which == 1) {
#pragma unroll
        for (int i = 0; i < 8; ++i) {
          int m = i + hi * 8;
          k[(((size_t)b * H_ + h) * T_ + t0 + m) * HD_ + n0 + cn] = (bf16)acc[i];
        }
      } else {
#pragma unroll
        for (int i = 0; i < 8; ++i) {
          int m = i + hi * 8;
          vt[(((size_t)b * H_ + h) * HD_ + n0 + cn) * T_ + t0 + m] = (bf16)acc[i];
        }
      }
    }
  }
}

// ---------------- fused causal flash attention ----------------
// grid B*H*(T/64), block 128 (4 waves); wave w handles row tile ti = 4*(blk%4)+w
__global__ __launch_bounds__(128) void attn_kernel(
    const bf16* __restrict__ q,   // [B,H,T,HD] (pre-scaled)
    const bf16* __restrict__ k,   // [B,H,T,HD]
    const bf16* __restrict__ vt,  // [B,H,HD,T]
    bf16* __restrict__ att)       // [B,T,D] (head-concat)
{
  __shared__ __align__(16) bf16 sp[4][16 * 32];   // per-wave P staging
  const int wave = threadIdx.x >> 5;
  const int lane = threadIdx.x & 31;
  const int cn = lane & 15, hi = lane >> 4;

  int gid = blockIdx.x;
  const int ti = (gid % (T_ / 64)) * 4 + wave;
  const int bh = gid / (T_ / 64);
  const int h = bh % H_, b = bh / H_;
  const int t0 = ti * 16;

  const bf16* qb = q  + ((size_t)b * H_ + h) * T_ * HD_;
  const bf16* kb = k  + ((size_t)b * H_ + h) * T_ * HD_;
  const bf16* vb = vt + ((size_t)b * H_ + h) * HD_ * T_;

  // Q tile as two 16x32 A fragments (K = head dim)
  v16bf aq0, aq1;
  {
    const bf16* qr = qb + (size_t)(t0 + cn) * HD_ + hi * 8;
    aq0 = make_v16(*(const v8bf*)(qr),      *(const v8bf*)(qr + 16));
    aq1 = make_v16(*(const v8bf*)(qr + 32), *(const v8bf*)(qr + 48));
  }

  float mrow[8], lrow[8];
  v8f o0 = {0.f,0.f,0.f,0.f,0.f,0.f,0.f,0.f}, o1 = o0, o2 = o0, o3 = o0;
#pragma unroll
  for (int i = 0; i < 8; ++i) { mrow[i] = -1e30f; lrow[i] = 0.f; }

  bf16* myp = sp[wave];
  const int nsteps = ti / 2 + 1;          // causal: cols [0, 32*nsteps)

  for (int st = 0; st < nsteps; ++st) {
    const int s0 = st * 32;

    // prefetch next step's K/V block into cache (global_prefetch_b8)
    if (st + 1 < nsteps) {
      __builtin_prefetch(kb + (size_t)(s0 + 32 + lane) * HD_, 0, 3);
      __builtin_prefetch(vb + (size_t)lane * T_ + s0 + 32, 0, 3);
      __builtin_prefetch(vb + (size_t)(32 + lane) * T_ + s0 + 32, 0, 3);
    }

    // S = Q * K^T for 32 columns (two 16x16 tiles, K-dim = 64 = 2 frags)
    v8f sA = {0.f,0.f,0.f,0.f,0.f,0.f,0.f,0.f}, sB = sA;
    sA = wmma_bf16(aq0, b_frag(kb + (size_t)s0 * HD_, HD_, 0),  sA);
    sA = wmma_bf16(aq1, b_frag(kb + (size_t)s0 * HD_, HD_, 32), sA);
    sB = wmma_bf16(aq0, b_frag(kb + (size_t)(s0 + 16) * HD_, HD_, 0),  sB);
    sB = wmma_bf16(aq1, b_frag(kb + (size_t)(s0 + 16) * HD_, HD_, 32), sB);

    // causal mask + online softmax stats (row = i + 8*hi, col = cn within tile)
    float sc[8];
#pragma unroll
    for (int i = 0; i < 8; ++i) {
      const int t = t0 + i + hi * 8;
      float sa = (s0 + cn      <= t) ? sA[i] : -1e30f;
      float sb = (s0 + 16 + cn <= t) ? sB[i] : -1e30f;
      float mx = half_max(fmaxf(sa, sb));
      float mnew = fmaxf(mrow[i], mx);
      sc[i] = __expf(mrow[i] - mnew);
      float ea = __expf(sa - mnew);
      float eb = __expf(sb - mnew);
      lrow[i] = lrow[i] * sc[i] + half_sum(ea + eb);
      mrow[i] = mnew;
      sA[i] = ea; sB[i] = eb;
      o0[i] *= sc[i]; o1[i] *= sc[i]; o2[i] *= sc[i]; o3[i] *= sc[i];
    }

    // C-layout -> A-layout transpose of P through per-wave LDS (16x32 bf16)
#pragma unroll
    for (int i = 0; i < 8; ++i) {
      int m = i + hi * 8;
      myp[m * 32 + cn]      = (bf16)sA[i];
      myp[m * 32 + 16 + cn] = (bf16)sB[i];
    }
    __builtin_amdgcn_wave_barrier();
    asm volatile("s_wait_dscnt 0" ::: "memory");   // same-wave DS in-order; data visible
    v16bf ap = a_frag(myp, 32, 0);
    __builtin_amdgcn_wave_barrier();

    // O += P * V  (B frags contiguous thanks to transposed V)
    o0 = wmma_bf16(ap, b_frag(vb + (size_t) 0 * T_ + s0, T_, 0), o0);
    o1 = wmma_bf16(ap, b_frag(vb + (size_t)16 * T_ + s0, T_, 0), o1);
    o2 = wmma_bf16(ap, b_frag(vb + (size_t)32 * T_ + s0, T_, 0), o2);
    o3 = wmma_bf16(ap, b_frag(vb + (size_t)48 * T_ + s0, T_, 0), o3);
  }

  // normalize and write concat output [B,T,H*HD]
#pragma unroll
  for (int i = 0; i < 8; ++i) {
    int m = i + hi * 8;
    float inv = 1.0f / lrow[i];
    bf16* dst = att + ((size_t)b * T_ + t0 + m) * D_ + h * HD_ + cn;
    dst[0]  = (bf16)(o0[i] * inv);
    dst[16] = (bf16)(o1[i] * inv);
    dst[32] = (bf16)(o2[i] * inv);
    dst[48] = (bf16)(o3[i] * inv);
  }
}

// ---------------- output projection: out = att @ Wp + bp ----------------
// grid B*T/16, block 128; wave w covers n-tiles {w, w+4, ...} of 24
__global__ __launch_bounds__(128) void proj_kernel(
    const bf16* __restrict__ att,  // [B,T,D]
    const bf16* __restrict__ Wpt,  // [D,D] transposed
    const float* __restrict__ bp,  // [D]
    float* __restrict__ out)       // [B,T,D]
{
  __shared__ __align__(16) bf16 sx[16 * D_];
  const int b  = blockIdx.x / (T_ / 16);
  const int ti = blockIdx.x % (T_ / 16);
  const int t0 = ti * 16;

  const bf16* src = att + ((size_t)b * T_ + t0) * D_;
#pragma unroll
  for (int e = threadIdx.x * 8; e < 16 * D_; e += 128 * 8)
    async_copy16(src + e, sx + e);
  wait_async0();
  __syncthreads();

  const int wave = threadIdx.x >> 5;
  const int lane = threadIdx.x & 31;
  const int cn = lane & 15, hi = lane >> 4;

  for (int nt = wave; nt < D_ / 16; nt += 4) {
    const int n0 = nt * 16;
    const bf16* wb = Wpt + (size_t)n0 * D_;
    v8f acc = {0.f,0.f,0.f,0.f,0.f,0.f,0.f,0.f};
#pragma unroll
    for (int kt = 0; kt < D_; kt += 32)
      acc = wmma_bf16(a_frag(sx, D_, kt), b_frag(wb, D_, kt), acc);
    const float bias = bp[n0 + cn];
#pragma unroll
    for (int i = 0; i < 8; ++i) {
      int m = i + hi * 8;
      out[((size_t)b * T_ + t0 + m) * D_ + n0 + cn] = acc[i] + bias;
    }
  }
}

// ---------------- host launcher ----------------
extern "C" void kernel_launch(void* const* d_in, const int* in_sizes, int n_in,
                              void* d_out, int out_size, void* d_ws, size_t ws_size,
                              hipStream_t stream) {
  const float* x  = (const float*)d_in[0];
  const float* Wq = (const float*)d_in[1];
  const float* Wk = (const float*)d_in[2];
  const float* Wv = (const float*)d_in[3];
  const float* Wp = (const float*)d_in[4];
  const float* bp = (const float*)d_in[5];
  float* out = (float*)d_out;

  const size_t nx   = (size_t)B_ * T_ * D_;       // 12.58M
  const size_t nqkv = (size_t)B_ * H_ * T_ * HD_; // 12.58M
  const size_t nw   = (size_t)H_ * D_ * HD_;
  const size_t nwp  = (size_t)D_ * D_;

  char* p = (char*)d_ws;
  bf16* xb  = (bf16*)p; p += nx   * sizeof(bf16);
  bf16* qd  = (bf16*)p; p += nqkv * sizeof(bf16);
  bf16* kd  = (bf16*)p; p += nqkv * sizeof(bf16);
  bf16* vtd = (bf16*)p; p += nqkv * sizeof(bf16);
  bf16* att = (bf16*)p; p += nx   * sizeof(bf16);
  bf16* Wqt = (bf16*)p; p += nw   * sizeof(bf16);
  bf16* Wkt = (bf16*)p; p += nw   * sizeof(bf16);
  bf16* Wvt = (bf16*)p; p += nw   * sizeof(bf16);
  bf16* Wpt = (bf16*)p; p += nwp  * sizeof(bf16);

  cvt_f32_bf16<<<(int)((nx + 255) / 256), 256, 0, stream>>>(x, xb, (int)nx);
  cvt_tr_w<<<(int)((nw + 255) / 256), 256, 0, stream>>>(Wq, Wqt);
  cvt_tr_w<<<(int)((nw + 255) / 256), 256, 0, stream>>>(Wk, Wkt);
  cvt_tr_w<<<(int)((nw + 255) / 256), 256, 0, stream>>>(Wv, Wvt);
  cvt_tr_wp<<<(int)((nwp + 255) / 256), 256, 0, stream>>>(Wp, Wpt);

  qkv_kernel<<<B_ * (T_ / 16), 128, 0, stream>>>(xb, Wqt, Wkt, Wvt, qd, kd, vtd);

  attn_kernel<<<B_ * H_ * (T_ / 64), 128, 0, stream>>>(qd, kd, vtd, att);

  proj_kernel<<<B_ * (T_ / 16), 128, 0, stream>>>(att, Wpt, bp, out);
}
